// MultiSequenceCrossAttention_72138270703627
// MI455X (gfx1250) — matrix operand order repaired and checked by reading, not compiled, not run on hardware
//
#include <hip/hip_runtime.h>
#include <hip/hip_bf16.h>

#define DINL __device__ __forceinline__

constexpr int BB   = 2;
constexpr int LQc  = 2048;
constexpr int LKc  = 2048;
constexpr int HIDc = 1024;
constexpr int NHc  = 16;
constexpr int HDc  = 64;

typedef __attribute__((ext_vector_type(16))) __bf16 bf16x16;
typedef __attribute__((ext_vector_type(8)))  float  v8f;

union Frag { bf16x16 v; unsigned u[8]; };

DINL unsigned short f2bf(float f) {
    unsigned u = __float_as_uint(f);
    unsigned r = (u + 0x7FFFu + ((u >> 16) & 1u)) >> 16;
    return (unsigned short)r;
}
DINL float bf2f(unsigned short h) { return __uint_as_float(((unsigned)h) << 16); }

DINL v8f wmma_bf16(const bf16x16 a, const bf16x16 b, v8f c) {
    // v_wmma_f32_16x16x32_bf16: D = A(16x32) * B(32x16) + C
    return __builtin_amdgcn_wmma_f32_16x16x32_bf16(false, a, false, b, (short)0, c, false, false);
}

// A-fragment K index (16-bit A, 16x32, ISA 7.12.2)
DINL int ka_idx(int v, int lh) { return ((v & 4) ? 16 : 0) + (lh ? 8 : 0) + ((v & 3) << 1); }
// B-fragment K index (16-bit B, 32x16)
DINL int kb_idx(int v, int lh) { return ((v & 4) ? 16 : 0) + (lh ? 8 : 0) + ((v & 3) << 1); }

// Generic LDS address -> 32-bit LDS byte offset (ISA 10.2: generic LDS addr is
// {SHARED_BASE[63:32], offset[31:0]}, hardware truncates to 32 bits).
DINL unsigned lds_off32(const void* p) { return (unsigned)(unsigned long long)p; }

// CDNA5 async DMA: global -> LDS, 16B per lane, tracked with ASYNCcnt.
DINL void async_b128_to_lds(unsigned lds_byte_off, const void* gaddr) {
    asm volatile("global_load_async_to_lds_b128 %0, %1, off"
                 :: "v"(lds_byte_off), "v"((unsigned long long)gaddr)
                 : "memory");
}
DINL void wait_asynccnt0() {
    asm volatile("s_wait_asynccnt 0" ::: "memory");
}

// ---------------------------------------------------------------------------
// Segment positions: cumsum of lengths + searchsorted(right)
// grid = 2*BB blocks (first BB = q mask, next BB = kv mask)
// ---------------------------------------------------------------------------
__global__ __launch_bounds__(256) void positions_kernel(
    const int* __restrict__ mask_q, const int* __restrict__ mask_kv,
    float* __restrict__ posq, int* __restrict__ valq,
    float* __restrict__ posk, int* __restrict__ segk, int* __restrict__ valk)
{
    const int which = blockIdx.x / BB;
    const int b = blockIdx.x % BB;
    const int L = which ? LKc : LQc;
    const int* lens = which ? (mask_kv + (size_t)b * LKc) : (mask_q + (size_t)b * LQc);
    __shared__ int bnd[2048];
    if (threadIdx.x == 0) {
        int sacc = 0;
        for (int i = 0; i < L; i++) { sacc += lens[i]; bnd[i] = sacc; }
    }
    __syncthreads();
    const int total = bnd[L - 1];
    for (int t = threadIdx.x; t < L; t += blockDim.x) {
        int lo = 0, hi = L;
        while (lo < hi) { int mid = (lo + hi) >> 1; if (bnd[mid] <= t) lo = mid + 1; else hi = mid; }
        int seg = lo < (L - 1) ? lo : (L - 1);
        int start = bnd[seg] - lens[seg];
        int valid = (t < total) ? 1 : 0;
        float pos = valid ? (float)(t - start) : 0.f;
        if (which == 0) {
            posq[(size_t)b * LQc + t] = pos;
            valq[(size_t)b * LQc + t] = valid;
        } else {
            posk[(size_t)b * LKc + t] = pos;
            segk[(size_t)b * LKc + t] = seg;
            valk[(size_t)b * LKc + t] = valid;
        }
    }
}

// ---------------------------------------------------------------------------
// Tiled WMMA GEMM: C[M,N] = (A[M,K] * B[K,N] + bias[N]) * scale
// Block 256 threads (8 waves), tile 128x128, K-step 32, software-pipelined:
// next tile is prefetched into registers (f32->bf16 folded into fetch) while
// the current LDS tile feeds the 8 WMMAs per wave.
// ---------------------------------------------------------------------------
template <bool ABF16, bool OUTBF16>
__global__ __launch_bounds__(256) void gemm_kernel(
    const void* __restrict__ Ap, int lda,
    const float* __restrict__ Bp, int ldb,
    const float* __restrict__ bias, float scale,
    void* __restrict__ Cp, int ldc,
    int M, int N, int K)
{
    constexpr int BM = 128, BN = 128, BK = 32;
    __shared__ unsigned short As[BM][BK + 8];
    __shared__ unsigned short Bs[BK][BN + 8];
    const int tid  = threadIdx.x;
    const int bm   = blockIdx.y * BM, bn = blockIdx.x * BN;
    const int lane = tid & 31, wave = tid >> 5;
    const int l15  = lane & 15, lh = lane >> 4;
    const int wm   = (wave & 1) * 64, wn = (wave >> 1) * 32;
    (void)M; (void)N;

    v8f acc[4][2] = {};
    uint2 aR[4], bR[4];

    auto fetch = [&](int k0) {
        #pragma unroll
        for (int i = 0; i < 4; i++) {
            int e = i * 256 + tid;
            {   // A tile 128x32
                int row = e >> 3, kq = (e & 7) * 4;
                if constexpr (ABF16) {
                    aR[i] = *(const uint2*)((const unsigned short*)Ap +
                                            (size_t)(bm + row) * lda + k0 + kq);
                } else {
                    float4 s4 = *(const float4*)((const float*)Ap +
                                                 (size_t)(bm + row) * lda + k0 + kq);
                    aR[i].x = (unsigned)f2bf(s4.x) | ((unsigned)f2bf(s4.y) << 16);
                    aR[i].y = (unsigned)f2bf(s4.z) | ((unsigned)f2bf(s4.w) << 16);
                }
            }
            {   // B tile 32x128
                int row = e >> 5, cq = (e & 31) * 4;
                float4 s4 = *(const float4*)(Bp + (size_t)(k0 + row) * ldb + bn + cq);
                bR[i].x = (unsigned)f2bf(s4.x) | ((unsigned)f2bf(s4.y) << 16);
                bR[i].y = (unsigned)f2bf(s4.z) | ((unsigned)f2bf(s4.w) << 16);
            }
        }
    };

    fetch(0);
    for (int k0 = 0; k0 < K; k0 += BK) {
        __syncthreads();
        #pragma unroll
        for (int i = 0; i < 4; i++) {
            int e = i * 256 + tid;
            int rowA = e >> 3, kq = (e & 7) * 4;
            *(uint2*)&As[rowA][kq] = aR[i];
            int rowB = e >> 5, cq = (e & 31) * 4;
            *(uint2*)&Bs[rowB][cq] = bR[i];
        }
        __syncthreads();
        if (k0 + BK < K) fetch(k0 + BK);   // overlap next global fetch with WMMA

        Frag a[4];
        #pragma unroll
        for (int mt = 0; mt < 4; mt++) {
            int mrow = wm + mt * 16 + l15;
            #pragma unroll
            for (int v = 0; v < 8; v++)
                a[mt].u[v] = *(const unsigned*)&As[mrow][ka_idx(v, lh)];
        }
        #pragma unroll
        for (int nt = 0; nt < 2; nt++) {
            Frag bf;
            int ncol = wn + nt * 16 + l15;
            #pragma unroll
            for (int v = 0; v < 8; v++) {
                int kk = kb_idx(v, lh);
                unsigned lo = Bs[kk][ncol], hi = Bs[kk + 1][ncol];
                bf.u[v] = lo | (hi << 16);
            }
            #pragma unroll
            for (int mt = 0; mt < 4; mt++)
                acc[mt][nt] = wmma_bf16(a[mt].v, bf.v, acc[mt][nt]);
        }
    }

    // Epilogue: C layout n = l15, m = r + 8*lh
    #pragma unroll
    for (int mt = 0; mt < 4; mt++) {
        #pragma unroll
        for (int nt = 0; nt < 2; nt++) {
            int col = bn + wn + nt * 16 + l15;
            float bv = bias ? bias[col] : 0.f;
            #pragma unroll
            for (int r = 0; r < 8; r++) {
                int row = bm + wm + mt * 16 + r + 8 * lh;
                float val = (acc[mt][nt][r] + bv) * scale;
                if constexpr (OUTBF16)
                    ((unsigned short*)Cp)[(size_t)row * ldc + col] = f2bf(val);
                else
                    ((float*)Cp)[(size_t)row * ldc + col] = val;
            }
        }
    }
}

// ---------------------------------------------------------------------------
// RoPE on Q (in place, bf16).  One thread per (token, head, j<32) pair.
// ---------------------------------------------------------------------------
__global__ __launch_bounds__(256) void ropeq_kernel(
    unsigned short* __restrict__ Qb, const float* __restrict__ posq)
{
    const long long idx = (long long)blockIdx.x * blockDim.x + threadIdx.x;
    const long long total = (long long)BB * LQc * NHc * (HDc / 2);
    if (idx >= total) return;
    const int j = (int)(idx & 31);
    const long long rest = idx >> 5;
    const int hh = (int)(rest & (NHc - 1));
    const long long tok = rest >> 4;            // b*LQ + t
    const float pos = posq[tok];
    unsigned short* base = Qb + tok * HIDc + hh * HDc;
    float x0 = bf2f(base[j]);
    float x1 = bf2f(base[j + 32]);
    float invf = __powf(10000.f, -(float)j * (1.f / 32.f));
    float sn, cs;
    __sincosf(pos * invf, &sn, &cs);
    base[j]      = f2bf(x0 * cs - x1 * sn);
    base[j + 32] = f2bf(x1 * cs + x0 * sn);
}

// ---------------------------------------------------------------------------
// RoPE + distance embedding on K (in place, bf16).
// ---------------------------------------------------------------------------
__global__ __launch_bounds__(256) void ropek_kernel(
    unsigned short* __restrict__ Kb,
    const float* __restrict__ posk, const int* __restrict__ segk,
    const int* __restrict__ valk, const float* __restrict__ dist)
{
    const long long idx = (long long)blockIdx.x * blockDim.x + threadIdx.x;
    const long long total = (long long)BB * LKc * NHc * (HDc / 2);
    if (idx >= total) return;
    const int j = (int)(idx & 31);
    const long long rest = idx >> 5;
    const int hh = (int)(rest & (NHc - 1));
    const long long tok = rest >> 4;            // b*LK + t
    const long long bidx = tok / LKc;
    const float pos = posk[tok];
    const float dd = valk[tok] ? dist[bidx * LKc + segk[tok]] : 0.f;
    unsigned short* base = Kb + tok * HIDc + hh * HDc;
    float x0 = bf2f(base[j]);
    float x1 = bf2f(base[j + 32]);
    float invf = __powf(10000.f, -(float)j * (1.f / 32.f));
    float sn, cs;
    __sincosf(pos * invf, &sn, &cs);
    // geomspace(1e-5, 0.25, 32): g_j = 1e-5 * 25000^(j/31)
    float g = 1.0e-5f * __powf(25000.0f, (float)j * (1.f / 31.f));
    float ang = dd * g;
    base[j]      = f2bf(x0 * cs - x1 * sn + __cosf(ang));
    base[j + 32] = f2bf(x1 * cs + x0 * sn + __sinf(ang));
}

// ---------------------------------------------------------------------------
// Flash attention: block = (128 q-rows, head h, batch b), 8 waves.
// Wave owns 16 q-rows.  K/V 64-token tiles are double-buffered in LDS and
// filled by CDNA5 async DMA (global_load_async_to_lds_b128 / ASYNCcnt), so
// the next tile streams in while WMMAs consume the current one.
// ---------------------------------------------------------------------------
__global__ __launch_bounds__(256) void attn_kernel(
    const unsigned short* __restrict__ Qb,
    const unsigned short* __restrict__ Kb,
    const unsigned short* __restrict__ Vb,
    const int* __restrict__ validq,
    const int* __restrict__ validk,
    unsigned short* __restrict__ Ob)
{
    const int b = blockIdx.z, h = blockIdx.y;
    const int tid = threadIdx.x, lane = tid & 31, wave = tid >> 5;
    const int l15 = lane & 15, lh = lane >> 4;
    const int qRow0 = blockIdx.x * 128 + wave * 16;

    __shared__ alignas(16) unsigned short Kt[2][64][72];   // [buf][token][dim]
    __shared__ alignas(16) unsigned short Vt[2][64][72];
    __shared__ alignas(16) unsigned short Pw[8][16][72];   // per-wave P staging

    const unsigned ktLds[2] = { lds_off32(&Kt[0][0][0]), lds_off32(&Kt[1][0][0]) };
    const unsigned vtLds[2] = { lds_off32(&Vt[0][0][0]), lds_off32(&Vt[1][0][0]) };

    auto issueTile = [&](int buf, int kt) {
        #pragma unroll
        for (int i = 0; i < 2; i++) {
            int e = i * 256 + tid;
            int row = e >> 3, c8 = (e & 7) * 8;
            size_t g = (size_t)(b * LKc + kt + row) * HIDc + h * HDc + c8;
            unsigned loff = (unsigned)((row * 72 + c8) * 2);
            async_b128_to_lds(ktLds[buf] + loff, Kb + g);
            async_b128_to_lds(vtLds[buf] + loff, Vb + g);
        }
    };

    // Q fragments: 16 rows x 64 dims -> two K=32 A-fragments
    Frag qf[2];
    #pragma unroll
    for (int half = 0; half < 2; half++) {
        #pragma unroll
        for (int v = 0; v < 8; v++) {
            int kk = half * 32 + ka_idx(v, lh);
            qf[half].u[v] = *(const unsigned*)(Qb +
                ((size_t)(b * LQc + qRow0 + l15) * HIDc + h * HDc + kk));
        }
    }

    float mrow[8], lrow[8];
    v8f o[4] = {};
    #pragma unroll
    for (int r = 0; r < 8; r++) { mrow[r] = -3.0e38f; lrow[r] = 0.f; }

    issueTile(0, 0);                    // prime the pipeline
    for (int kt = 0; kt < LKc; kt += 64) {
        const int cur = (kt >> 6) & 1;
        wait_asynccnt0();               // my async DMAs (into buffer cur) done
        __syncthreads();                // everyone's DMAs done; prev reads done
        if (kt + 64 < LKc) issueTile(1 - cur, kt + 64);   // stream next tile

        // S = Q * K^T : 4 tiles of 16(q) x 16(k-tokens)
        v8f s[4];
        float maskadd[4];
        #pragma unroll
        for (int nt = 0; nt < 4; nt++) {
            int tok = nt * 16 + l15;
            Frag kb0, kb1;
            #pragma unroll
            for (int v = 0; v < 8; v++) {
                int kk = kb_idx(v, lh);
                kb0.u[v] = *(const unsigned*)&Kt[cur][tok][kk];   // K^T pairs contiguous
                kb1.u[v] = *(const unsigned*)&Kt[cur][tok][kk + 32];
            }
            v8f accv = {};
            accv = wmma_bf16(qf[0].v, kb0.v, accv);
            accv = wmma_bf16(qf[1].v, kb1.v, accv);
            s[nt] = accv;
            maskadd[nt] = validk[(size_t)b * LKc + kt + tok] ? 0.f : -1.0e9f;
        }
        #pragma unroll
        for (int nt = 0; nt < 4; nt++) {
            #pragma unroll
            for (int r = 0; r < 8; r++) s[nt][r] += maskadd[nt];
        }

        // Online softmax: row m = r + 8*lh; xor-shuffles <16 stay in half-wave
        #pragma unroll
        for (int r = 0; r < 8; r++) {
            float mx = fmaxf(fmaxf(s[0][r], s[1][r]), fmaxf(s[2][r], s[3][r]));
            #pragma unroll
            for (int d = 1; d < 16; d <<= 1) mx = fmaxf(mx, __shfl_xor(mx, d, 32));
            float mnew = fmaxf(mrow[r], mx);
            float corr = __expf(mrow[r] - mnew);
            mrow[r] = mnew;
            float rs = 0.f;
            #pragma unroll
            for (int nt = 0; nt < 4; nt++) {
                float p = __expf(s[nt][r] - mnew);
                s[nt][r] = p;
                rs += p;
            }
            #pragma unroll
            for (int d = 1; d < 16; d <<= 1) rs += __shfl_xor(rs, d, 32);
            lrow[r] = lrow[r] * corr + rs;
            #pragma unroll
            for (int dt = 0; dt < 4; dt++) o[dt][r] *= corr;
        }

        // C-layout -> A-layout via per-wave LDS
        #pragma unroll
        for (int nt = 0; nt < 4; nt++) {
            #pragma unroll
            for (int r = 0; r < 8; r++)
                Pw[wave][r + 8 * lh][nt * 16 + l15] = f2bf(s[nt][r]);
        }
        __builtin_amdgcn_wave_barrier();
        asm volatile("s_wait_dscnt 0" ::: "memory");
        __builtin_amdgcn_wave_barrier();

        Frag pf[2];
        #pragma unroll
        for (int half = 0; half < 2; half++) {
            #pragma unroll
            for (int v = 0; v < 8; v++) {
                int kk = half * 32 + ka_idx(v, lh);
                pf[half].u[v] = *(const unsigned*)&Pw[wave][l15][kk];
            }
        }
        // O += P (16x64) * V (64x64): 4 head-dim tiles
        #pragma unroll
        for (int dt = 0; dt < 4; dt++) {
            Frag vf0, vf1;
            int dcol = dt * 16 + l15;
            #pragma unroll
            for (int v = 0; v < 8; v++) {
                int kk = kb_idx(v, lh);
                vf0.u[v] = (unsigned)Vt[cur][kk][dcol]      | ((unsigned)Vt[cur][kk + 1][dcol] << 16);
                vf1.u[v] = (unsigned)Vt[cur][kk + 32][dcol] | ((unsigned)Vt[cur][kk + 33][dcol] << 16);
            }
            o[dt] = wmma_bf16(pf[0].v, vf0.v, o[dt]);
            o[dt] = wmma_bf16(pf[1].v, vf1.v, o[dt]);
        }
    }

    // Normalize, apply valid_q, write bf16 (b, q, h*64 + d)
    #pragma unroll
    for (int r = 0; r < 8; r++) {
        int qrow = qRow0 + r + 8 * lh;
        float inv = lrow[r] > 0.f ? 1.f / lrow[r] : 0.f;
        float fac = validq[(size_t)b * LQc + qrow] ? inv : 0.f;
        #pragma unroll
        for (int dt = 0; dt < 4; dt++)
            Ob[(size_t)(b * LQc + qrow) * HIDc + h * HDc + dt * 16 + l15] =
                f2bf(o[dt][r] * fac);
    }
}

// ---------------------------------------------------------------------------
extern "C" void kernel_launch(void* const* d_in, const int* in_sizes, int n_in,
                              void* d_out, int out_size, void* d_ws, size_t ws_size,
                              hipStream_t stream)
{
    (void)in_sizes; (void)n_in; (void)out_size; (void)ws_size;
    const float* q_states  = (const float*)d_in[0];
    const float* kv_states = (const float*)d_in[1];
    const float* distances = (const float*)d_in[2];
    const int*   mask_q    = (const int*)d_in[3];
    const int*   mask_kv   = (const int*)d_in[4];
    const float* Wq  = (const float*)d_in[5];
    const float* bq  = (const float*)d_in[6];
    const float* Wkv = (const float*)d_in[7];
    const float* bkv = (const float*)d_in[8];
    const float* Wo  = (const float*)d_in[9];
    const float* bo  = (const float*)d_in[10];
    float* out = (float*)d_out;

    char* w = (char*)d_ws;
    const size_t nQ = (size_t)BB * LQc * HIDc;           // 4M elements
    unsigned short* Qb = (unsigned short*)w; w += nQ * 2;
    unsigned short* Kb = (unsigned short*)w; w += nQ * 2;
    unsigned short* Vb = (unsigned short*)w; w += nQ * 2;
    unsigned short* Ob = (unsigned short*)w; w += nQ * 2;
    float* posq = (float*)w; w += (size_t)BB * LQc * 4;
    int*   valq = (int*)w;   w += (size_t)BB * LQc * 4;
    float* posk = (float*)w; w += (size_t)BB * LKc * 4;
    int*   segk = (int*)w;   w += (size_t)BB * LKc * 4;
    int*   valk = (int*)w;   w += (size_t)BB * LKc * 4;

    positions_kernel<<<2 * BB, 256, 0, stream>>>(mask_q, mask_kv,
                                                 posq, valq, posk, segk, valk);

    const int M = BB * LQc, N = HIDc, K = HIDc;
    dim3 gg(N / 128, M / 128);
    // Q = (x*Wq + bq) * hd^-0.5
    gemm_kernel<false, true><<<gg, 256, 0, stream>>>(
        q_states, HIDc, Wq, HIDc, bq, 0.125f, Qb, HIDc, M, N, K);
    // K = x*Wkv[:, :HID] + bkv[:HID]
    gemm_kernel<false, true><<<gg, 256, 0, stream>>>(
        kv_states, HIDc, Wkv, 2 * HIDc, bkv, 1.0f, Kb, HIDc, M, N, K);
    // V = x*Wkv[:, HID:] + bkv[HID:]
    gemm_kernel<false, true><<<gg, 256, 0, stream>>>(
        kv_states, HIDc, Wkv + HIDc, 2 * HIDc, bkv + HIDc, 1.0f, Vb, HIDc, M, N, K);

    const long long nRope = (long long)BB * LQc * NHc * (HDc / 2);
    ropeq_kernel<<<(unsigned)((nRope + 255) / 256), 256, 0, stream>>>(Qb, posq);
    ropek_kernel<<<(unsigned)((nRope + 255) / 256), 256, 0, stream>>>(
        Kb, posk, segk, valk, distances);

    attn_kernel<<<dim3(LQc / 128, NHc, BB), 256, 0, stream>>>(
        Qb, Kb, Vb, valq, valk, Ob);

    // out = O*Wo + bo (f32 output)
    gemm_kernel<true, false><<<gg, 256, 0, stream>>>(
        Ob, HIDc, Wo, HIDc, bo, 1.0f, out, HIDc, M, N, K);
}
